// W2VHuBERT_Quantizier_51977694216303
// MI455X (gfx1250) — matrix operand order repaired and checked by reading, not compiled, not run on hardware
//
#include <hip/hip_runtime.h>
#include <hip/hip_bf16.h>

// W2V/HuBERT vector quantizer: argmin_k || e - c_k || over K=1024 clusters.
// argmin of (||c||^2 - 2 e.c) == reference argmax(-dist).  GEMM via
// v_wmma_f32_16x16x32_f16 with a 3-term f16 hi/lo split (~f32 accuracy).
//
// Fast path: clusters pre-converted ONCE into WMMA B-fragment layout
// (f16 hi/lo) + ||c||^2 table in d_ws; main loop is wmma + b128 loads only.
// Fallback path (small ws): convert clusters on the fly per block.

#define DD    768                // feature dim
#define KK    1024               // clusters
#define RPB   64                 // rows per block (4 x 16-row WMMA tiles)
#define TILES 4
#define KB    (DD / 32)          // 24 k-blocks of 32
#define NPASS (KK / (8 * 16))    // 8 passes: 8 waves x 16 cols each
#define NBLKS (KK / 16)          // 64 cluster column blocks
#define FRAG_HALVES ((size_t)NBLKS * KB * 32 * 16)  // halves per hi (or lo) array

typedef _Float16 v16h __attribute__((ext_vector_type(16)));
typedef float    v8f  __attribute__((ext_vector_type(8)));

// ---------------------------------------------------------------------------
// Prep 1: clusters -> pre-swizzled B fragments, f16 hi/lo.
// Fragment layout: [nblk][kb][lane][16 halves]; lane: n = L&15, hf = L>>4,
// halves cover K = kb*32 + 16*hf .. +16 (contiguous k order; A staged the same way).
// ---------------------------------------------------------------------------
__global__ __launch_bounds__(256)
void vq_prep_frags(const float* __restrict__ clus,
                   _Float16* __restrict__ fragHi,
                   _Float16* __restrict__ fragLo)
{
    int gid  = blockIdx.x * 256 + threadIdx.x;    // [0, NBLKS*KB*32)
    int lane = gid & 31;
    int kb   = (gid >> 5) % KB;
    int nblk = gid / (KB * 32);
    int n    = nblk * 16 + (lane & 15);
    int hf   = lane >> 4;

    const float4* p4 = (const float4*)(clus + (size_t)n * DD + kb * 32 + 16 * hf);
    float4 f0 = p4[0], f1 = p4[1], f2 = p4[2], f3 = p4[3];
    float xs[16] = {f0.x, f0.y, f0.z, f0.w, f1.x, f1.y, f1.z, f1.w,
                    f2.x, f2.y, f2.z, f2.w, f3.x, f3.y, f3.z, f3.w};
    v16h vh, vl;
#pragma unroll
    for (int i = 0; i < 16; ++i) {
        float x = xs[i];
        _Float16 h = (_Float16)x;
        vh[i] = h;
        vl[i] = (_Float16)(x - (float)h);
    }
    size_t off = (size_t)gid * 16;
    *(v16h*)(fragHi + off) = vh;
    *(v16h*)(fragLo + off) = vl;
}

// ---------------------------------------------------------------------------
// Prep 2: ||c_k||^2 table (one wave per cluster, 8 clusters per block).
// ---------------------------------------------------------------------------
__global__ __launch_bounds__(256)
void vq_prep_sq(const float* __restrict__ clus, float* __restrict__ sqArr)
{
    int lane = threadIdx.x & 31;
    int wv   = threadIdx.x >> 5;
    int n    = blockIdx.x * 8 + wv;
    const float* row = clus + (size_t)n * DD;
    float s = 0.f;
    for (int i = lane; i < DD; i += 32) s += row[i] * row[i];
#pragma unroll
    for (int off = 16; off >= 1; off >>= 1) s += __shfl_xor(s, off, 32);
    if (lane == 0) sqArr[n] = s;
}

// ---------------------------------------------------------------------------
// Shared device helper: stage A (embeddings) into LDS in fragment order.
// pair (kb, lane, j): m = lane&15, hf = lane>>4,
// K = kb*32 + 16*hf + 2j, elems {K, K+1}
// ---------------------------------------------------------------------------
__device__ __forceinline__
void stage_A(const float* embBase, _Float16* sAhi, _Float16* sAlo, int tid)
{
    for (int p = tid; p < TILES * KB * 256; p += 256) {
        int t  = p / (KB * 256);
        int q  = p - t * (KB * 256);
        int kb = q >> 8;
        int ln = (q >> 3) & 31;
        int j  = q & 7;
        int m  = ln & 15;
        int hf = ln >> 4;
        int kB = kb * 32 + 16 * hf + 2 * j;
        const float* src = embBase + (size_t)(t * 16 + m) * DD + kB;
        float x0 = src[0];
        float x1 = src[1];
        _Float16 h0 = (_Float16)x0, h1 = (_Float16)x1;
        _Float16 l0 = (_Float16)(x0 - (float)h0);
        _Float16 l1 = (_Float16)(x1 - (float)h1);
        int off = t * (KB * 512) + kb * 512 + ln * 16 + j * 2;
        sAhi[off] = h0; sAhi[off + 1] = h1;
        sAlo[off] = l0; sAlo[off + 1] = l1;
    }
}

// ---------------------------------------------------------------------------
// Shared device helper: wave/block argmin reduction + output.
// ---------------------------------------------------------------------------
__device__ __forceinline__
void reduce_and_store(float bestS[TILES][8], int bestI[TILES][8],
                      float (*rS)[RPB], int (*rI)[RPB],
                      int tid, int wv, int hf, int nLoc,
                      float* out, size_t row0)
{
#pragma unroll
    for (int t = 0; t < TILES; ++t)
#pragma unroll
        for (int r = 0; r < 8; ++r) {
#pragma unroll
            for (int off = 8; off >= 1; off >>= 1) {
                float os = __shfl_xor(bestS[t][r], off, 32);
                int   oi = __shfl_xor(bestI[t][r], off, 32);
                if (os < bestS[t][r] ||
                    (os == bestS[t][r] && oi < bestI[t][r])) {
                    bestS[t][r] = os; bestI[t][r] = oi;
                }
            }
        }
    // half 0 (lane 0) holds rows m=r; half 1 (lane 16) holds rows m=8+r
    if (nLoc == 0) {
#pragma unroll
        for (int t = 0; t < TILES; ++t)
#pragma unroll
            for (int r = 0; r < 8; ++r) {
                int m = t * 16 + hf * 8 + r;
                rS[wv][m] = bestS[t][r];
                rI[wv][m] = bestI[t][r];
            }
    }
    __syncthreads();
    if (tid < RPB) {
        float bs = rS[0][tid];
        int   bi = rI[0][tid];
#pragma unroll
        for (int w2 = 1; w2 < 8; ++w2) {
            float s  = rS[w2][tid];
            int   i2 = rI[w2][tid];
            if (s < bs || (s == bs && i2 < bi)) { bs = s; bi = i2; }
        }
        out[row0 + tid] = (float)bi;   // integer index stored as float
    }
}

// ---------------------------------------------------------------------------
// FAST main kernel: B fragments + ||c||^2 preconverted in workspace.
// ---------------------------------------------------------------------------
__global__ __launch_bounds__(256, 1)
void vq_argmin_wmma_pre(const float* __restrict__ emb,
                        const _Float16* __restrict__ fragHi,
                        const _Float16* __restrict__ fragLo,
                        const float* __restrict__ sqArr,
                        float* __restrict__ out)
{
    __shared__ __align__(32) _Float16 sAhi[TILES * KB * 512];
    __shared__ __align__(32) _Float16 sAlo[TILES * KB * 512];
    __shared__ float rS[8][RPB];
    __shared__ int   rI[8][RPB];

    const int tid = threadIdx.x;
    const size_t row0 = (size_t)blockIdx.x * RPB;
    stage_A(emb + row0 * DD, sAhi, sAlo, tid);
    __syncthreads();

    const int lane = tid & 31;
    const int wv   = tid >> 5;
    const int nLoc = lane & 15;
    const int hf   = lane >> 4;

    float bestS[TILES][8];
    int   bestI[TILES][8];
#pragma unroll
    for (int t = 0; t < TILES; ++t)
#pragma unroll
        for (int r = 0; r < 8; ++r) { bestS[t][r] = 3.0e38f; bestI[t][r] = 0; }

    for (int pass = 0; pass < NPASS; ++pass) {
        const int n0   = pass * 128 + wv * 16;
        const int nblk = pass * 8 + wv;
        const size_t fbase = (((size_t)nblk * KB) * 32 + lane) * 16;
        const _Float16* fh = fragHi + fbase;
        const _Float16* fl = fragLo + fbase;
        const float sqn = sqArr[n0 + nLoc];

        v8f acc[TILES];
#pragma unroll
        for (int t = 0; t < TILES; ++t)
            acc[t] = (v8f){0.f, 0.f, 0.f, 0.f, 0.f, 0.f, 0.f, 0.f};

#pragma unroll 2
        for (int kb = 0; kb < KB; ++kb) {
            v16h bh = *(const v16h*)(fh + (size_t)kb * 512);
            v16h bl = *(const v16h*)(fl + (size_t)kb * 512);
#pragma unroll
            for (int t = 0; t < TILES; ++t) {
                const int aoff = t * (KB * 512) + kb * 512 + lane * 16;
                v16h ah = *(const v16h*)&sAhi[aoff];
                v16h al = *(const v16h*)&sAlo[aoff];
                acc[t] = __builtin_amdgcn_wmma_f32_16x16x32_f16(
                    false, ah, false, bh, (short)0, acc[t], false, false);
                acc[t] = __builtin_amdgcn_wmma_f32_16x16x32_f16(
                    false, ah, false, bl, (short)0, acc[t], false, false);
                acc[t] = __builtin_amdgcn_wmma_f32_16x16x32_f16(
                    false, al, false, bh, (short)0, acc[t], false, false);
            }
        }

        const int myN = n0 + nLoc;
#pragma unroll
        for (int t = 0; t < TILES; ++t)
#pragma unroll
            for (int r = 0; r < 8; ++r) {
                float s = sqn - 2.0f * acc[t][r];
                if (s < bestS[t][r]) { bestS[t][r] = s; bestI[t][r] = myN; }
            }
    }

    reduce_and_store(bestS, bestI, rS, rI, tid, wv, hf, nLoc, out, row0);
}

// ---------------------------------------------------------------------------
// FALLBACK main kernel (no workspace): convert B on the fly.
// ---------------------------------------------------------------------------
__global__ __launch_bounds__(256, 1)
void vq_argmin_wmma_otf(const float* __restrict__ emb,
                        const float* __restrict__ clus,
                        float* __restrict__ out)
{
    __shared__ __align__(32) _Float16 sAhi[TILES * KB * 512];
    __shared__ __align__(32) _Float16 sAlo[TILES * KB * 512];
    __shared__ float rS[8][RPB];
    __shared__ int   rI[8][RPB];

    const int tid = threadIdx.x;
    const size_t row0 = (size_t)blockIdx.x * RPB;
    stage_A(emb + row0 * DD, sAhi, sAlo, tid);
    __syncthreads();

    const int lane = tid & 31;
    const int wv   = tid >> 5;
    const int nLoc = lane & 15;
    const int hf   = lane >> 4;

    float bestS[TILES][8];
    int   bestI[TILES][8];
#pragma unroll
    for (int t = 0; t < TILES; ++t)
#pragma unroll
        for (int r = 0; r < 8; ++r) { bestS[t][r] = 3.0e38f; bestI[t][r] = 0; }

    for (int pass = 0; pass < NPASS; ++pass) {
        const int n0 = pass * 128 + wv * 16;
        const float* crow = clus + (size_t)(n0 + nLoc) * DD + 16 * hf;

        v8f acc[TILES];
#pragma unroll
        for (int t = 0; t < TILES; ++t)
            acc[t] = (v8f){0.f, 0.f, 0.f, 0.f, 0.f, 0.f, 0.f, 0.f};
        float sq = 0.f;

        for (int kb = 0; kb < KB; ++kb) {
            const float4* p4 = (const float4*)(crow + kb * 32);
            float4 f0 = p4[0], f1 = p4[1], f2 = p4[2], f3 = p4[3];
            float xs[16] = {f0.x, f0.y, f0.z, f0.w, f1.x, f1.y, f1.z, f1.w,
                            f2.x, f2.y, f2.z, f2.w, f3.x, f3.y, f3.z, f3.w};
            v16h bh, bl;
#pragma unroll
            for (int i = 0; i < 16; ++i) {
                float x = xs[i];
                _Float16 h = (_Float16)x;
                bh[i] = h;
                bl[i] = (_Float16)(x - (float)h);
                sq += x * x;
            }
#pragma unroll
            for (int t = 0; t < TILES; ++t) {
                const int aoff = t * (KB * 512) + kb * 512 + lane * 16;
                v16h ah = *(const v16h*)&sAhi[aoff];
                v16h al = *(const v16h*)&sAlo[aoff];
                acc[t] = __builtin_amdgcn_wmma_f32_16x16x32_f16(
                    false, ah, false, bh, (short)0, acc[t], false, false);
                acc[t] = __builtin_amdgcn_wmma_f32_16x16x32_f16(
                    false, ah, false, bl, (short)0, acc[t], false, false);
                acc[t] = __builtin_amdgcn_wmma_f32_16x16x32_f16(
                    false, al, false, bh, (short)0, acc[t], false, false);
            }
        }
        sq += __shfl_xor(sq, 16, 32);

        const int myN = n0 + nLoc;
#pragma unroll
        for (int t = 0; t < TILES; ++t)
#pragma unroll
            for (int r = 0; r < 8; ++r) {
                float s = sq - 2.0f * acc[t][r];
                if (s < bestS[t][r]) { bestS[t][r] = s; bestI[t][r] = myN; }
            }
    }

    reduce_and_store(bestS, bestI, rS, rI, tid, wv, hf, nLoc, out, row0);
}

// ---------------------------------------------------------------------------
extern "C" void kernel_launch(void* const* d_in, const int* in_sizes, int n_in,
                              void* d_out, int out_size, void* d_ws, size_t ws_size,
                              hipStream_t stream) {
    (void)in_sizes; (void)n_in; (void)out_size;
    const float* emb  = (const float*)d_in[0];   // [32,1500,768] f32
    const float* clus = (const float*)d_in[1];   // [1024,768]    f32
    float* out = (float*)d_out;                  // [32*1500] indices as float

    const int M = 32 * 1500;                     // 48000 rows, 64 per block
    dim3 grid(M / RPB), block(256);

    const size_t needWs = FRAG_HALVES * 2 * sizeof(_Float16) + KK * sizeof(float);
    if (d_ws != nullptr && ws_size >= needWs) {
        _Float16* fragHi = (_Float16*)d_ws;
        _Float16* fragLo = fragHi + FRAG_HALVES;
        float*    sqArr  = (float*)((char*)d_ws + FRAG_HALVES * 2 * sizeof(_Float16));
        hipLaunchKernelGGL(vq_prep_frags, dim3(NBLKS * KB * 32 / 256), dim3(256),
                           0, stream, clus, fragHi, fragLo);
        hipLaunchKernelGGL(vq_prep_sq, dim3(KK / 8), dim3(256),
                           0, stream, clus, sqArr);
        hipLaunchKernelGGL(vq_argmin_wmma_pre, grid, block, 0, stream,
                           emb, fragHi, fragLo, sqArr, out);
    } else {
        hipLaunchKernelGGL(vq_argmin_wmma_otf, grid, block, 0, stream,
                           emb, clus, out);
    }
}
